// BiDirectionalMinGRU_5626407158119
// MI455X (gfx1250) — matrix operand order repaired and checked by reading, not compiled, not run on hardware
//
#include <hip/hip_runtime.h>
#include <stdint.h>

// ---------------------------------------------------------------------------
// BiDirectionalMinGRU for MI455X (gfx1250).
// All GEMM operands pre-swizzled into WMMA fragment-major layout so the GEMM
// inner loops are pure {global_load_b128 x N, v_wmma} with no LDS / barriers.
// K-loops fully unrolled so accumulators stay in one fixed register set
// (no loop-carried copies -> no WMMA->VALU hazard NOPs).
// ---------------------------------------------------------------------------

#define BB    16
#define LL    2048
#define HH    512
#define NTT   8
#define OUTD  1032
#define KPAD  1056
#define HEADH 256
#define ROWS  (BB * LL)        // 32768
#define CHUNK 64
#define NCH   (LL / CHUNK)     // 32
#define KT_G  (HH / 32)        // 16 K-tiles, gate GEMM
#define KT_H  (KPAD / 32)      // 33 K-tiles, head GEMM

typedef __bf16 bf16;
typedef bf16     v16bf __attribute__((ext_vector_type(16)));
typedef float    v8f   __attribute__((ext_vector_type(8)));
typedef uint16_t v16u  __attribute__((ext_vector_type(16)));

// float -> bf16 bits, round-to-nearest-even
__device__ __forceinline__ uint16_t f2bf(float f) {
  uint32_t u = __builtin_bit_cast(uint32_t, f);
  u += 0x7FFFu + ((u >> 16) & 1u);
  return (uint16_t)(u >> 16);
}

__device__ __forceinline__ v16bf load_frag(const uint16_t* p) {
  return __builtin_bit_cast(v16bf, *(const v16u*)p);
}

// Branch-free activations (v_exp_f32 based).
__device__ __forceinline__ float sigmoid_fast(float x) {
  return 1.f / (1.f + __expf(-x));
}
__device__ __forceinline__ float tanh_fast(float x) {
  const float e = __expf(-2.f * fabsf(x));
  const float t = (1.f - e) / (1.f + e);
  return copysignf(t, x);
}

// --- Fragment-major swizzle maps (per 16x32 A tile / 32x16 B tile, 512 elems,
//     each lane's 16 contiguous bf16 at lane*16). ---
// A (MxK): lane = m + (kk>=16?16:0) ; e = kk&15
__device__ __forceinline__ size_t a_swz_off(int row, int k, int ktiles) {
  const int m = row & 15, kk = k & 31;
  const int lane = m + ((kk & 16) ? 16 : 0);
  const int e = kk & 15;
  return ((size_t)(row >> 4) * ktiles + (k >> 5)) * 512 + lane * 16 + e;
}
// B (KxN): lane = n + (kk>=16?16:0) ; e = kk&15
__device__ __forceinline__ size_t b_swz_off(int k, int n, int ntiles) {
  const int nn = n & 15, kk = k & 31;
  const int lane = nn + ((kk & 16) ? 16 : 0);
  const int e = kk & 15;
  return ((size_t)(k >> 5) * ntiles + (n >> 4)) * 512 + lane * 16 + e;
}

// ---------------------------------------------------------------------------
// Phase 0: t-MLP + input projections (K=10, VALU fp32) -> swizzled bf16 A.
// ---------------------------------------------------------------------------
__global__ __launch_bounds__(256) void prep_kernel(
    const float* __restrict__ x, const float* __restrict__ t,
    const float* __restrict__ Wt1, const float* __restrict__ bt1,
    const float* __restrict__ Wt2, const float* __restrict__ bt2,
    const float* __restrict__ Wpf, const float* __restrict__ bpf,
    const float* __restrict__ Wpb, const float* __restrict__ bpb,
    float* __restrict__ t_enc, uint16_t* __restrict__ inp_f,
    uint16_t* __restrict__ inp_b) {
  const int row = blockIdx.x;
  const int tid = threadIdx.x;
  const float tt = t[row];
  float xc[10];
  xc[0] = x[row * 2 + 0];
  xc[1] = x[row * 2 + 1];
  float e1[8];
#pragma unroll
  for (int j = 0; j < 8; ++j) e1[j] = fmaxf(tt * Wt1[j] + bt1[j], 0.f);
#pragma unroll
  for (int j = 0; j < 8; ++j) {
    float s = bt2[j];
#pragma unroll
    for (int i = 0; i < 8; ++i) s += e1[i] * Wt2[i * 8 + j];
    xc[2 + j] = s;
  }
  if (tid < 8) t_enc[row * 8 + tid] = xc[2 + tid];
#pragma unroll
  for (int rep = 0; rep < 2; ++rep) {
    const int h = tid + rep * 256;
    float sf = bpf[h], sb = bpb[h];
#pragma unroll
    for (int i = 0; i < 10; ++i) {
      sf += xc[i] * Wpf[i * HH + h];
      sb += xc[i] * Wpb[i * HH + h];
    }
    const size_t off = a_swz_off(row, h, KT_G);
    inp_f[off] = f2bf(sf);
    inp_b[off] = f2bf(sb);
  }
}

// ---------------------------------------------------------------------------
// Weight conversions: fp32 row-major -> bf16 fragment-major.
// ---------------------------------------------------------------------------
__global__ __launch_bounds__(256) void cvt_w_swz_kernel(
    const float* __restrict__ src, uint16_t* __restrict__ dst) {
  const int i = blockIdx.x * 256 + threadIdx.x;  // 512*512
  const int k = i >> 9, n = i & (HH - 1);
  dst[b_swz_off(k, n, HH / 16)] = f2bf(src[i]);
}

__global__ __launch_bounds__(256) void cvt_wg1_swz_kernel(
    const float* __restrict__ src, uint16_t* __restrict__ dst) {
  const int k = blockIdx.x;    // 0..1055
  const int n = threadIdx.x;   // 0..255
  const uint16_t v = (k < OUTD) ? f2bf(src[(size_t)k * HEADH + n]) : (uint16_t)0;
  dst[b_swz_off(k, n, HEADH / 16)] = v;
}

// ---------------------------------------------------------------------------
// Phase 2: fused dual GEMM per direction; wave tile 16(M) x 64(N) x {Wz,Wh}
// => 8 WMMAs per K-step per wave. No LDS, no barriers; K fully unrolled.
// ---------------------------------------------------------------------------
__global__ __launch_bounds__(256) void gate_gemm_kernel(
    const uint16_t* __restrict__ Asw, const uint16_t* __restrict__ Bzsw,
    const uint16_t* __restrict__ Bhsw, const float* __restrict__ bz,
    const float* __restrict__ bh, float* __restrict__ a_out,
    float* __restrict__ b_out) {
  const int lane = threadIdx.x & 31;
  const int wave = threadIdx.x >> 5;
  const int rowTile = blockIdx.x * 8 + wave;  // 0..2047
  const int colTile0 = blockIdx.y * 4;        // 0..28

  v8f accz[4], acch[4];
#pragma unroll
  for (int j = 0; j < 4; ++j) {
    accz[j] = (v8f){0.f, 0.f, 0.f, 0.f, 0.f, 0.f, 0.f, 0.f};
    acch[j] = (v8f){0.f, 0.f, 0.f, 0.f, 0.f, 0.f, 0.f, 0.f};
  }

  const uint16_t* pa = Asw + (size_t)rowTile * KT_G * 512 + lane * 16;
  const uint16_t* pbz = Bzsw + (size_t)colTile0 * 512 + lane * 16;
  const uint16_t* pbh = Bhsw + (size_t)colTile0 * 512 + lane * 16;
  const size_t bstep = (size_t)(HH / 16) * 512;  // next K-tile of B

#pragma unroll
  for (int kt = 0; kt < KT_G; ++kt) {
    const v16bf a = load_frag(pa + (size_t)kt * 512);
    __builtin_prefetch(pa + (size_t)(kt + 1) * 512, 0, 3);  // speculative, safe
#pragma unroll
    for (int j = 0; j < 4; ++j) {
      const v16bf vz = load_frag(pbz + (size_t)kt * bstep + (size_t)j * 512);
      const v16bf vh = load_frag(pbh + (size_t)kt * bstep + (size_t)j * 512);
      accz[j] = __builtin_amdgcn_wmma_f32_16x16x32_bf16(false, a, false, vz,
                                                        (short)0, accz[j],
                                                        false, false);
      acch[j] = __builtin_amdgcn_wmma_f32_16x16x32_bf16(false, a, false, vh,
                                                        (short)0, acch[j],
                                                        false, false);
    }
  }

  // C layout: VGPR r -> M = r + (lane<16 ? 0 : 8), N = lane&15
  const int rtop = rowTile * 16 + ((lane < 16) ? 0 : 8);
#pragma unroll
  for (int j = 0; j < 4; ++j) {
    const int col = (colTile0 + j) * 16 + (lane & 15);
    const float bzc = bz[col], bhc = bh[col];
#pragma unroll
    for (int r = 0; r < 8; ++r) {
      const int row = rtop + r;
      const float zg = sigmoid_fast(accz[j][r] + bzc);
      const float ht = tanh_fast(acch[j][r] + bhc);
      a_out[(size_t)row * HH + col] = 1.f - zg;
      b_out[(size_t)row * HH + col] = zg * ht;
    }
  }
}

// ---------------------------------------------------------------------------
// Phase 3: chunked parallel scan (affine recurrence composition, 3 passes).
// Forward:  h_fwd[p] = S_p,  S_0 = 0,      S_{p+1} = a[p]*S_p + b[p]
// Backward: h_bwd[p] = U_{L-1-p}, U_0 = 0, U_{q+1} = a[L-1-q]*U_q + b[L-1-q]
// ---------------------------------------------------------------------------
__device__ __forceinline__ void scan_decode(int g, int& h, int& c, int& b,
                                            int& d) {
  h = g & (HH - 1);
  c = (g >> 9) & (NCH - 1);
  b = (g >> 14) & (BB - 1);
  d = g >> 18;
}

__global__ __launch_bounds__(256) void scan1_kernel(
    const float* __restrict__ af, const float* __restrict__ bf,
    const float* __restrict__ ab2, const float* __restrict__ bb2,
    float* __restrict__ P, float* __restrict__ Q) {
  const int g = blockIdx.x * 256 + threadIdx.x;
  int h, c, b, dir;
  scan_decode(g, h, c, b, dir);
  const float* A = dir ? ab2 : af;
  const float* Bv = dir ? bb2 : bf;
  float Pv = 1.f, Qv = 0.f;
  for (int s = 0; s < CHUNK; ++s) {
    const int q = c * CHUNK + s;
    const int p = dir ? (LL - 1 - q) : q;
    const size_t idx = ((size_t)(b * LL + p)) * HH + h;
    const float a = A[idx], bb = Bv[idx];
    Qv = a * Qv + bb;
    Pv = a * Pv;
  }
  P[g] = Pv;
  Q[g] = Qv;
}

__global__ __launch_bounds__(256) void scan2_kernel(
    const float* __restrict__ P, const float* __restrict__ Q,
    float* __restrict__ Sst) {
  const int g = blockIdx.x * 256 + threadIdx.x;  // 2*16*512
  const int h = g & (HH - 1);
  const int b = (g >> 9) & (BB - 1);
  const int dir = g >> 13;
  float S = 0.f;
  for (int c = 0; c < NCH; ++c) {
    const size_t idx = (((size_t)(dir * BB + b)) * NCH + c) * HH + h;
    Sst[idx] = S;
    S = P[idx] * S + Q[idx];
  }
}

__global__ __launch_bounds__(256) void scan3_kernel(
    const float* __restrict__ af, const float* __restrict__ bf,
    const float* __restrict__ ab2, const float* __restrict__ bb2,
    const float* __restrict__ Sst, float* __restrict__ hfw,
    float* __restrict__ hbw) {
  const int g = blockIdx.x * 256 + threadIdx.x;
  int h, c, b, dir;
  scan_decode(g, h, c, b, dir);
  const float* A = dir ? ab2 : af;
  const float* Bv = dir ? bb2 : bf;
  float* Hout = dir ? hbw : hfw;
  float S = Sst[g];
  for (int s = 0; s < CHUNK; ++s) {
    const int q = c * CHUNK + s;
    const int p = dir ? (LL - 1 - q) : q;
    const size_t idx = ((size_t)(b * LL + p)) * HH + h;
    Hout[idx] = S;
    S = A[idx] * S + Bv[idx];
  }
}

// ---------------------------------------------------------------------------
// Phase 4: fused LayerNorm over [h_fwd|h_bwd|t_enc] + time_scale, emitted as
// fragment-major bf16 rows zero-padded K 1032 -> 1056.
// ---------------------------------------------------------------------------
__device__ __forceinline__ float ln_val(const float* hf, const float* hb,
                                        const float* te, int row, int j) {
  if (j < HH) return hf[(size_t)row * HH + j];
  if (j < 2 * HH) return hb[(size_t)row * HH + (j - HH)];
  return te[(size_t)row * NTT + (j - 2 * HH)];
}

__global__ __launch_bounds__(256) void ln_kernel(
    const float* __restrict__ hf, const float* __restrict__ hb,
    const float* __restrict__ te, const float* __restrict__ ln_g,
    const float* __restrict__ ln_b, const float* __restrict__ tscale,
    uint16_t* __restrict__ hbi) {
  const int row = blockIdx.x;
  const int tid = threadIdx.x;
  __shared__ float r1[256], r2[256];
  float s = 0.f, ss = 0.f;
  for (int j = tid; j < OUTD; j += 256) {
    const float v = ln_val(hf, hb, te, row, j);
    s += v;
    ss += v * v;
  }
  r1[tid] = s;
  r2[tid] = ss;
  __syncthreads();
  for (int st = 128; st > 0; st >>= 1) {
    if (tid < st) {
      r1[tid] += r1[tid + st];
      r2[tid] += r2[tid + st];
    }
    __syncthreads();
  }
  const float mu = r1[0] * (1.f / OUTD);
  const float var = r2[0] * (1.f / OUTD) - mu * mu;
  const float inv = rsqrtf(var + 1e-5f);
  const float tsc = tscale[0];
  for (int j = tid; j < KPAD; j += 256) {
    uint16_t o = 0;
    if (j < OUTD) {
      float y = (ln_val(hf, hb, te, row, j) - mu) * inv * ln_g[j] + ln_b[j];
      if (j >= 2 * HH) y *= tsc;
      o = f2bf(y);
    }
    hbi[a_swz_off(row, j, KT_H)] = o;
  }
}

// ---------------------------------------------------------------------------
// Phase 5: head GEMM (M=32768, N=256, K=1056), exact-GELU epilogue.
// Wave tile 16x64, 4 WMMAs per K-step, no LDS / barriers; K fully unrolled.
// ---------------------------------------------------------------------------
__global__ __launch_bounds__(256) void head_gemm_kernel(
    const uint16_t* __restrict__ Asw, const uint16_t* __restrict__ Bsw,
    const float* __restrict__ bias, float* __restrict__ outp) {
  const int lane = threadIdx.x & 31;
  const int wave = threadIdx.x >> 5;
  const int rowTile = blockIdx.x * 8 + wave;  // 0..2047
  const int colTile0 = blockIdx.y * 4;        // 0..12

  v8f acc[4];
#pragma unroll
  for (int j = 0; j < 4; ++j)
    acc[j] = (v8f){0.f, 0.f, 0.f, 0.f, 0.f, 0.f, 0.f, 0.f};

  const uint16_t* pa = Asw + (size_t)rowTile * KT_H * 512 + lane * 16;
  const uint16_t* pb = Bsw + (size_t)colTile0 * 512 + lane * 16;
  const size_t bstep = (size_t)(HEADH / 16) * 512;

#pragma unroll
  for (int kt = 0; kt < KT_H; ++kt) {
    const v16bf a = load_frag(pa + (size_t)kt * 512);
    __builtin_prefetch(pa + (size_t)(kt + 1) * 512, 0, 3);  // speculative, safe
#pragma unroll
    for (int j = 0; j < 4; ++j) {
      const v16bf vb = load_frag(pb + (size_t)kt * bstep + (size_t)j * 512);
      acc[j] = __builtin_amdgcn_wmma_f32_16x16x32_bf16(false, a, false, vb,
                                                       (short)0, acc[j],
                                                       false, false);
    }
  }

  const int rtop = rowTile * 16 + ((lane < 16) ? 0 : 8);
#pragma unroll
  for (int j = 0; j < 4; ++j) {
    const int col = (colTile0 + j) * 16 + (lane & 15);
    const float bc = bias[col];
#pragma unroll
    for (int r = 0; r < 8; ++r) {
      const float v = acc[j][r] + bc;
      const float g = 0.5f * v * (1.f + erff(v * 0.70710678118654752f));
      outp[(size_t)(rtop + r) * HEADH + col] = g;
    }
  }
}

// ---------------------------------------------------------------------------
// Phase 6: out[row] = hidden_row . Wg2 + bg2  (one wave per row)
// ---------------------------------------------------------------------------
__global__ __launch_bounds__(256) void head_out_kernel(
    const float* __restrict__ hid, const float* __restrict__ Wg2,
    const float* __restrict__ bg2, float* __restrict__ outp) {
  const int gwave = (blockIdx.x * 256 + threadIdx.x) >> 5;
  const int lane = threadIdx.x & 31;
  float s = 0.f;
#pragma unroll
  for (int i = 0; i < 8; ++i) {
    const int n = lane + 32 * i;
    s += hid[(size_t)gwave * HEADH + n] * Wg2[n];
  }
#pragma unroll
  for (int off = 16; off > 0; off >>= 1) s += __shfl_xor(s, off, 32);
  if (lane == 0) outp[gwave] = s + bg2[0];
}

// ---------------------------------------------------------------------------
extern "C" void kernel_launch(void* const* d_in, const int* in_sizes, int n_in,
                              void* d_out, int out_size, void* d_ws,
                              size_t ws_size, hipStream_t stream) {
  (void)in_sizes; (void)n_in; (void)out_size; (void)ws_size;
  const float* x    = (const float*)d_in[0];
  const float* t    = (const float*)d_in[1];
  const float* Wt1  = (const float*)d_in[2];
  const float* bt1  = (const float*)d_in[3];
  const float* Wt2  = (const float*)d_in[4];
  const float* bt2  = (const float*)d_in[5];
  const float* Wpf  = (const float*)d_in[6];
  const float* bpf  = (const float*)d_in[7];
  const float* Wpb  = (const float*)d_in[8];
  const float* bpb  = (const float*)d_in[9];
  const float* Wzf  = (const float*)d_in[10];
  const float* bzf  = (const float*)d_in[11];
  const float* Whf  = (const float*)d_in[12];
  const float* bhf  = (const float*)d_in[13];
  const float* Wzb  = (const float*)d_in[14];
  const float* bzb  = (const float*)d_in[15];
  const float* Whb  = (const float*)d_in[16];
  const float* bhb  = (const float*)d_in[17];
  const float* ln_g = (const float*)d_in[18];
  const float* ln_b = (const float*)d_in[19];
  const float* tsc  = (const float*)d_in[20];
  const float* Wg1  = (const float*)d_in[21];
  const float* bg1  = (const float*)d_in[22];
  const float* Wg2  = (const float*)d_in[23];
  const float* bg2  = (const float*)d_in[24];
  float* outp = (float*)d_out;

  // Workspace carve-up (region sizes are multiples of 256B).
  char* base = (char*)d_ws;
  size_t off = 0;
  auto take = [&](size_t bytes) -> void* {
    void* p = base + off;
    off += (bytes + 255) & ~(size_t)255;
    return p;
  };
  float*    t_enc = (float*)take((size_t)ROWS * NTT * 4);
  uint16_t* Wzf_b = (uint16_t*)take((size_t)HH * HH * 2);
  uint16_t* Whf_b = (uint16_t*)take((size_t)HH * HH * 2);
  uint16_t* Wzb_b = (uint16_t*)take((size_t)HH * HH * 2);
  uint16_t* Whb_b = (uint16_t*)take((size_t)HH * HH * 2);
  uint16_t* Wg1_b = (uint16_t*)take((size_t)KPAD * HEADH * 2);
  const size_t nPQ = (size_t)2 * BB * NCH * HH;  // 524288
  float* Pbuf = (float*)take(nPQ * 4);
  float* Qbuf = (float*)take(nPQ * 4);
  float* Sst  = (float*)take(nPQ * 4);
  uint16_t* inp_f = (uint16_t*)take((size_t)ROWS * HH * 2);
  uint16_t* inp_b = (uint16_t*)take((size_t)ROWS * HH * 2);
  float* af  = (float*)take((size_t)ROWS * HH * 4);
  float* bfv = (float*)take((size_t)ROWS * HH * 4);
  float* ab2 = (float*)take((size_t)ROWS * HH * 4);
  float* bb2 = (float*)take((size_t)ROWS * HH * 4);
  float* hfw = (float*)take((size_t)ROWS * HH * 4);
  float* hbw = (float*)take((size_t)ROWS * HH * 4);
  // Aliases (lifetimes disjoint on the single stream):
  uint16_t* hbi = (uint16_t*)af;   // 66MB into af+bfv (128MB), dead after scan3
  float*    hid = (float*)inp_f;   // 32MB into inp_f (32MB), dead after GEMMs

  // 0) t-MLP + input projections (-> swizzled bf16 A operands)
  prep_kernel<<<ROWS, 256, 0, stream>>>(x, t, Wt1, bt1, Wt2, bt2, Wpf, bpf,
                                        Wpb, bpb, t_enc, inp_f, inp_b);
  // 1) weight conversions (fp32 -> fragment-major bf16)
  const int nW = HH * HH;
  cvt_w_swz_kernel<<<nW / 256, 256, 0, stream>>>(Wzf, Wzf_b);
  cvt_w_swz_kernel<<<nW / 256, 256, 0, stream>>>(Whf, Whf_b);
  cvt_w_swz_kernel<<<nW / 256, 256, 0, stream>>>(Wzb, Wzb_b);
  cvt_w_swz_kernel<<<nW / 256, 256, 0, stream>>>(Whb, Whb_b);
  cvt_wg1_swz_kernel<<<KPAD, 256, 0, stream>>>(Wg1, Wg1_b);
  // 2) gate GEMMs (dual-B WMMA, LDS-free) per direction
  gate_gemm_kernel<<<dim3(ROWS / 128, HH / 64), 256, 0, stream>>>(
      inp_f, Wzf_b, Whf_b, bzf, bhf, af, bfv);
  gate_gemm_kernel<<<dim3(ROWS / 128, HH / 64), 256, 0, stream>>>(
      inp_b, Wzb_b, Whb_b, bzb, bhb, ab2, bb2);
  // 3) chunked parallel scan (both directions)
  scan1_kernel<<<(unsigned)(nPQ / 256), 256, 0, stream>>>(af, bfv, ab2, bb2,
                                                          Pbuf, Qbuf);
  scan2_kernel<<<(2 * BB * HH) / 256, 256, 0, stream>>>(Pbuf, Qbuf, Sst);
  scan3_kernel<<<(unsigned)(nPQ / 256), 256, 0, stream>>>(af, bfv, ab2, bb2,
                                                          Sst, hfw, hbw);
  // 4) fused LayerNorm + time_scale -> swizzled padded bf16 rows
  ln_kernel<<<ROWS, 256, 0, stream>>>(hfw, hbw, t_enc, ln_g, ln_b, tsc, hbi);
  // 5) head GEMM + exact GELU
  head_gemm_kernel<<<dim3(ROWS / 128, HEADH / 64), 256, 0, stream>>>(
      hbi, Wg1_b, bg1, hid);
  // 6) final projection (wave-per-row shuffle reduction)
  head_out_kernel<<<ROWS / 8, 256, 0, stream>>>(hid, Wg2, bg2, outp);
}
